// DeepSeekMoe_69054484185324
// MI455X (gfx1250) — matrix operand order repaired and checked by reading, not compiled, not run on hardware
//
#include <hip/hip_runtime.h>
#include <hip/hip_bf16.h>

// ---------------- problem constants (from reference setup) ----------------
#define DIM     1024          // model dim (d_in == d_out)
#define TOKENS  4096          // B*S = 2*2048
#define NR      16            // routed experts
#define MTILE   32            // per-block M tile (2 x 16-row WMMA frags)
#define NTILE   256           // per-block N tile: 8 waves * 32
#define WAVE_N  32            // per-wave N tile (2 x 16x16 WMMA tiles)

typedef __attribute__((ext_vector_type(16))) __bf16 v16bf;
typedef __attribute__((ext_vector_type(8)))  __bf16 v8bf;
typedef __attribute__((ext_vector_type(4)))  __bf16 v4bf;
typedef __attribute__((ext_vector_type(8)))  float  v8f;
typedef __attribute__((ext_vector_type(4)))  float  v4f;

// ---------------- bf16 fragment loaders (wave32 WMMA layouts) ----------------
// A (16x32 bf16): lane l<16 -> row l, K {k0..k0+7, k0+16..k0+23}
//                 lane l>=16 -> row l-16, K {k0+8..k0+15, k0+24..k0+31}
__device__ __forceinline__ v16bf load_a_frag(const __bf16* __restrict__ row,
                                             int k0, int half) {
    const __bf16* p = row + k0 + half * 8;
    v8bf lo = *(const v8bf*)(p);        // 16B load
    v8bf hi = *(const v8bf*)(p + 16);   // 16B load
    v16bf a;
#pragma unroll
    for (int i = 0; i < 8; ++i) { a[i] = lo[i]; a[8 + i] = hi[i]; }
    return a;
}

// B (32x16 bf16): lane l holds column N=l&15; lanes 0-15 K {k0..k0+7, k0+16..k0+23},
// lanes 16-31 K {k0+8..k0+15, k0+24..k0+31}.  Column N of B == row N of W (u @ W^T),
// i.e. two 8-bf16 runs of one W row.
__device__ __forceinline__ v16bf load_b_frag(const __bf16* __restrict__ wrow,
                                             int k0, int half) {
    const __bf16* p = wrow + k0 + half * 8;
    v8bf lo = *(const v8bf*)(p);
    v8bf hi = *(const v8bf*)(p + 16);
    v16bf b;
#pragma unroll
    for (int i = 0; i < 8; ++i) { b[i] = lo[i]; b[8 + i] = hi[i]; }
    return b;
}

#define WMMA_BF16(A, B, C) \
    __builtin_amdgcn_wmma_f32_16x16x32_bf16(false, (A), false, (B), (short)0, (C), false, false)

// ---------------- prep: fp32 -> bf16 bulk conversion (one-time, L2-resident) --
__global__ void cvt_bf16_bulk(const float* __restrict__ src,
                              __bf16* __restrict__ dst) {
    const size_t i = (size_t)(blockIdx.x * blockDim.x + threadIdx.x) * 4;
    v4f v = *(const v4f*)(src + i);
    v4bf o;
#pragma unroll
    for (int j = 0; j < 4; ++j) o[j] = (__bf16)v[j];
    *(v4bf*)(dst + i) = o;
}

// fold the 2 shared experts into one bf16 matrix + bias; zero bucket counters
__global__ void moe_prep(const float* __restrict__ Ws, const float* __restrict__ bs,
                         __bf16* __restrict__ Wsum, float* __restrict__ bsum,
                         int* __restrict__ counts) {
    const size_t i = (size_t)(blockIdx.x * blockDim.x + threadIdx.x) * 4;
    v4f a = *(const v4f*)(Ws + i);
    v4f b = *(const v4f*)(Ws + i + (size_t)DIM * DIM);
    v4bf o;
#pragma unroll
    for (int j = 0; j < 4; ++j) o[j] = (__bf16)(0.5f * (a[j] + b[j]));
    *(v4bf*)(Wsum + i) = o;
    const int t = blockIdx.x * blockDim.x + threadIdx.x;
    if (t < DIM / 4) {
        v4f x = *(const v4f*)(bs + t * 4);
        v4f y = *(const v4f*)(bs + DIM + t * 4);
        v4f z;
#pragma unroll
        for (int j = 0; j < 4; ++j) z[j] = 0.5f * (x[j] + y[j]);
        *(v4f*)(bsum + t * 4) = z;
    }
    if (t < NR) counts[t] = 0;
}

// ---------------- routing: scores -> softmax -> top-2 -> expert buckets ------
__global__ void moe_route(const float* __restrict__ u, const float* __restrict__ cent,
                          const float* __restrict__ eb,
                          int* __restrict__ counts, int* __restrict__ lists,
                          float* __restrict__ gates) {
    const int token = blockIdx.x;
    const int tid = threadIdx.x;               // 256 threads = 16 experts x 16 slices
    const int e = tid >> 4, j = tid & 15;
    const float* ur = u + (size_t)token * DIM;
    const float* cr = cent + (size_t)e * DIM;
    float p = 0.f;
    for (int d = j; d < DIM; d += 16) p += ur[d] * cr[d];
    __shared__ float red[NR][16];
    __shared__ float score[NR];
    red[e][j] = p;
    __syncthreads();
    if (j == 0) {
        float s = 0.f;
#pragma unroll
        for (int t = 0; t < 16; ++t) s += red[e][t];
        score[e] = s + eb[e];
    }
    __syncthreads();
    if (tid == 0) {
        float mx = score[0];
#pragma unroll
        for (int t = 1; t < NR; ++t) mx = fmaxf(mx, score[t]);
        float sm[NR]; float sum = 0.f;
#pragma unroll
        for (int t = 0; t < NR; ++t) { sm[t] = expf(score[t] - mx); sum += sm[t]; }
        const float inv = 1.f / sum;
        int i0 = 0; float v0 = -1.f;
#pragma unroll
        for (int t = 0; t < NR; ++t) if (sm[t] > v0) { v0 = sm[t]; i0 = t; }
        int i1 = -1; float v1 = -1.f;
#pragma unroll
        for (int t = 0; t < NR; ++t) if (t != i0 && sm[t] > v1) { v1 = sm[t]; i1 = t; }
        int s0 = atomicAdd(&counts[i0], 1);
        lists[i0 * TOKENS + s0] = token; gates[i0 * TOKENS + s0] = v0 * inv;
        int s1 = atomicAdd(&counts[i1], 1);
        lists[i1 * TOKENS + s1] = token; gates[i1 * TOKENS + s1] = v1 * inv;
    }
}

// ------------- shared experts + residual: out = u + u@Wsum^T + bsum ----------
// Block: 32 tokens x 256 cols.  Wave: 32x32 via 2x2 WMMA tiles.
__global__ void moe_shared_gemm(const float* __restrict__ u,
                                const __bf16* __restrict__ ubf,
                                const __bf16* __restrict__ Wsum,
                                const float* __restrict__ bsum,
                                float* __restrict__ out) {
    const int lane = threadIdx.x & 31;
    const int wave = threadIdx.x >> 5;
    const int r = lane & 15, half = lane >> 4;
    const int m0 = blockIdx.y * MTILE;
    const int n0 = blockIdx.x * NTILE + wave * WAVE_N;
    const __bf16* arow0 = ubf + (size_t)(m0 + r) * DIM;
    const __bf16* arow1 = arow0 + (size_t)16 * DIM;
    const __bf16* wrow0 = Wsum + (size_t)(n0 + r) * DIM;
    const __bf16* wrow1 = wrow0 + (size_t)16 * DIM;
    v8f c00 = {}, c01 = {}, c10 = {}, c11 = {};
#pragma unroll 2
    for (int k0 = 0; k0 < DIM; k0 += 32) {
        v16bf a0 = load_a_frag(arow0, k0, half);
        v16bf a1 = load_a_frag(arow1, k0, half);
        v16bf b0 = load_b_frag(wrow0, k0, half);
        v16bf b1 = load_b_frag(wrow1, k0, half);
        c00 = WMMA_BF16(a0, b0, c00);
        c01 = WMMA_BF16(a0, b1, c01);
        c10 = WMMA_BF16(a1, b0, c10);
        c11 = WMMA_BF16(a1, b1, c11);
    }
    const int n = n0 + r;
    const float bn0 = bsum[n];
    const float bn1 = bsum[n + 16];
#pragma unroll
    for (int rr = 0; rr < 8; ++rr) {
        const int mA = m0 + rr + 8 * half;
        const size_t oA = (size_t)mA * DIM + n;
        out[oA]      = u[oA]      + c00[rr] + bn0;
        out[oA + 16] = u[oA + 16] + c01[rr] + bn1;
        const size_t oB = oA + (size_t)16 * DIM;
        out[oB]      = u[oB]      + c10[rr] + bn0;
        out[oB + 16] = u[oB + 16] + c11[rr] + bn1;
    }
}

// ------------- routed experts: grouped GEMM over buckets, gated atomics ------
__global__ void moe_routed_gemm(const __bf16* __restrict__ ubf,
                                const __bf16* __restrict__ Wrbf,
                                const float* __restrict__ br,
                                const int* __restrict__ counts,
                                const int* __restrict__ lists,
                                const float* __restrict__ gates,
                                float* __restrict__ out) {
    const int e = blockIdx.z;
    const int cnt = counts[e];
    const int tile = blockIdx.y;
    if (tile * MTILE >= cnt) return;           // uniform per block: safe early exit

    __shared__ int   stok[MTILE];
    __shared__ float sg[MTILE];
    if (threadIdx.x < MTILE) {
        const int slot = tile * MTILE + threadIdx.x;
        if (slot < cnt) {
            stok[threadIdx.x] = lists[e * TOKENS + slot];
            sg[threadIdx.x]   = gates[e * TOKENS + slot];
        } else {                               // pad rows: valid token, zero gate
            stok[threadIdx.x] = lists[e * TOKENS];
            sg[threadIdx.x]   = 0.f;
        }
    }
    __syncthreads();

    const int lane = threadIdx.x & 31;
    const int wave = threadIdx.x >> 5;
    const int r = lane & 15, half = lane >> 4;
    const int n0 = blockIdx.x * NTILE + wave * WAVE_N;
    const __bf16* arow0 = ubf + (size_t)stok[r] * DIM;
    const __bf16* arow1 = ubf + (size_t)stok[16 + r] * DIM;
    const __bf16* wrow0 = Wrbf + (size_t)e * DIM * DIM + (size_t)(n0 + r) * DIM;
    const __bf16* wrow1 = wrow0 + (size_t)16 * DIM;
    v8f c00 = {}, c01 = {}, c10 = {}, c11 = {};
#pragma unroll 2
    for (int k0 = 0; k0 < DIM; k0 += 32) {
        v16bf a0 = load_a_frag(arow0, k0, half);
        v16bf a1 = load_a_frag(arow1, k0, half);
        v16bf b0 = load_b_frag(wrow0, k0, half);
        v16bf b1 = load_b_frag(wrow1, k0, half);
        c00 = WMMA_BF16(a0, b0, c00);
        c01 = WMMA_BF16(a0, b1, c01);
        c10 = WMMA_BF16(a1, b0, c10);
        c11 = WMMA_BF16(a1, b1, c11);
    }
    const int n = n0 + r;
    const float bn0 = br[e * DIM + n];
    const float bn1 = br[e * DIM + n + 16];
#pragma unroll
    for (int rr = 0; rr < 8; ++rr) {
        const int mA = rr + 8 * half;          // rows 0..15 of tile
        const int mB = mA + 16;                // rows 16..31 of tile
        const float gA = sg[mA];
        const float gB = sg[mB];
        const size_t oA = (size_t)stok[mA] * DIM + n;
        const size_t oB = (size_t)stok[mB] * DIM + n;
        atomicAdd(&out[oA],      gA * (c00[rr] + bn0));  // token in <=2 buckets
        atomicAdd(&out[oA + 16], gA * (c01[rr] + bn1));
        atomicAdd(&out[oB],      gB * (c10[rr] + bn0));
        atomicAdd(&out[oB + 16], gB * (c11[rr] + bn1));
    }
}

// ---------------- host-side launch ----------------
extern "C" void kernel_launch(void* const* d_in, const int* in_sizes, int n_in,
                              void* d_out, int out_size, void* d_ws, size_t ws_size,
                              hipStream_t stream) {
    const float* u    = (const float*)d_in[0];   // [2,2048,1024]
    const float* cent = (const float*)d_in[1];   // [16,1024]
    const float* eb   = (const float*)d_in[2];   // [16]
    const float* Wr   = (const float*)d_in[3];   // [16,1024,1024]
    const float* br   = (const float*)d_in[4];   // [16,1024]
    const float* Ws   = (const float*)d_in[5];   // [2,1024,1024]
    const float* bs   = (const float*)d_in[6];   // [2,1024]
    float* out = (float*)d_out;

    char* ws = (char*)d_ws;
    size_t off = 0;
    int*    counts = (int*)(ws + off);   off += 1024;
    int*    lists  = (int*)(ws + off);   off += (size_t)NR * TOKENS * 4;
    float*  gates  = (float*)(ws + off); off += (size_t)NR * TOKENS * 4;
    float*  bsum   = (float*)(ws + off); off += DIM * 4;
    __bf16* ubf    = (__bf16*)(ws + off); off += (size_t)TOKENS * DIM * 2;
    __bf16* Wsumbf = (__bf16*)(ws + off); off += (size_t)DIM * DIM * 2;
    __bf16* Wrbf   = (__bf16*)(ws + off); off += (size_t)NR * DIM * DIM * 2;

    // 1) one-time fp32->bf16 conversions (stay L2-resident afterwards)
    cvt_bf16_bulk<<<(TOKENS * DIM / 4) / 256, 256, 0, stream>>>(u, ubf);
    cvt_bf16_bulk<<<(NR * DIM * DIM / 4) / 256, 256, 0, stream>>>(Wr, Wrbf);
    moe_prep<<<(DIM * DIM / 4) / 256, 256, 0, stream>>>(Ws, bs, Wsumbf, bsum, counts);
    // 2) routing: scores, softmax, top-2 -> per-expert buckets
    moe_route<<<TOKENS, 256, 0, stream>>>(u, cent, eb, counts, lists, gates);
    // 3) out = u + shared-mean (dense WMMA GEMM, bf16 inputs, f32 accum)
    moe_shared_gemm<<<dim3(DIM / NTILE, TOKENS / MTILE), 256, 0, stream>>>(
        u, ubf, Wsumbf, bsum, out);
    // 4) out += gated routed experts (grouped WMMA GEMM over buckets)
    moe_routed_gemm<<<dim3(DIM / NTILE, TOKENS / MTILE, NR), 256, 0, stream>>>(
        ubf, Wrbf, br, counts, lists, gates, out);
    (void)in_sizes; (void)n_in; (void)out_size; (void)ws_size;
}
